// AttnBlock_30803505447522
// MI455X (gfx1250) — matrix operand order repaired and checked by reading, not compiled, not run on hardware
//
#include <hip/hip_runtime.h>
#include <hip/hip_bf16.h>

// AttnBlock (GN -> QKV -> full spatial attention -> out-proj + residual)
// MI455X / gfx1250, wave32, bf16 WMMA (v_wmma_f32_16x16x32_bf16), f32 accum.
//
// b=8, c=256, n=h*w=4096.  ~155 GFLOP vs ~100MB HBM traffic -> compute bound;
// bf16 WMMA (8192 MAC/instr) is 8x the f32 WMMA path, so all GEMMs run bf16
// with f32 accumulation; softmax / GN / residual stay f32.
//
// Softmax scale (c^-0.5) AND log2(e) are folded into Q at projection time, so
// the flash-attention inner loop computes p = exp2(S - m) with zero extra
// multiplies on the WMMA accumulators.

typedef __bf16 bf16_t;
typedef __attribute__((ext_vector_type(16))) __bf16 v16bf;
typedef __attribute__((ext_vector_type(8)))  __bf16 v8bf;
typedef __attribute__((ext_vector_type(8)))  float  v8f;

#define C_DIM 256
#define N_SP  4096   // 64*64

union AB16 { v16bf v; v8bf h[2]; };

// A-fragment (16x32 MxK bf16): lane holds row m=lane&15; element e maps to
// k = hi*16 + e.  Caller passes rowbase already offset by hi*16.
__device__ inline v16bf load_frag_a(const bf16_t* rowbase, int k0) {
  AB16 u;
  u.h[0] = *(const v8bf*)(rowbase + k0);
  u.h[1] = *(const v8bf*)(rowbase + k0 + 8);
  return u.v;
}

__device__ inline v8f wmma_bf16(v16bf a, v16bf b, v8f c) {
  return __builtin_amdgcn_wmma_f32_16x16x32_bf16(
      /*neg_a=*/false, a, /*neg_b=*/false, b,
      /*c_mod=*/(short)0, c, /*reuse_a=*/false, /*reuse_b=*/false);
}

// ---------------------------------------------------------------- weights->bf16
__global__ void wcvt_kernel(const float* __restrict__ w, bf16_t* __restrict__ o, int n) {
  int i = blockIdx.x * 256 + threadIdx.x;
  if (i < n) o[i] = (bf16_t)w[i];
}

// ---------------------------------------------------------------- GroupNorm
// One block per (b,g).  Two passes: LDS-tree mean/var, then normalize and
// store hn transposed to [b, i, c] bf16 (c contiguous -> WMMA fragment loads
// later are per-lane contiguous 16B/32B chunks).
__global__ __launch_bounds__(256) void gn_kernel(const float* __restrict__ x,
                                                 const float* __restrict__ gsc,
                                                 const float* __restrict__ gbi,
                                                 bf16_t* __restrict__ hn) {
  __shared__ float red[256];
  __shared__ float sred[256];
  int bg = blockIdx.x;
  int b = bg >> 3, g = bg & 7;
  const float* base = x + ((size_t)b * C_DIM + (size_t)g * 32) * N_SP;
  int tid = threadIdx.x;
  float s = 0.f, ss = 0.f;
  for (int idx = tid; idx < 32 * N_SP; idx += 256) {
    float v = base[idx];
    s += v; ss += v * v;
  }
  red[tid] = s; sred[tid] = ss;
  __syncthreads();
  for (int off = 128; off > 0; off >>= 1) {
    if (tid < off) { red[tid] += red[tid + off]; sred[tid] += sred[tid + off]; }
    __syncthreads();
  }
  const float invN = 1.0f / (32.0f * (float)N_SP);
  float mean = red[0] * invN;
  float var  = sred[0] * invN - mean * mean;
  float rstd = rsqrtf(var + 1e-6f);

  int cl = tid & 31, ii = tid >> 5;
  int c = g * 32 + cl;
  float scl = gsc[c] * rstd;
  float sh  = gbi[c] - mean * scl;
  bf16_t* hb = hn + (size_t)b * N_SP * C_DIM;
  const float* xr = base + (size_t)cl * N_SP;
  for (int i = ii; i < N_SP; i += 8)
    hb[(size_t)i * C_DIM + c] = (bf16_t)(xr[i] * scl + sh);
}

// ---------------------------------------------------------------- QKV proj
// Per wave: one 16-row i strip.
//  q,k (transposed form): M=i, N=o, K=ci.  A = hn tiles, B = weight rows
//    (wq[o][ci] row-major IS B-layout-contiguous).  D stored [b,i,c] coalesced.
//    q is pre-scaled by log2(e)/sqrt(c) -> attention S is in exp2 units.
//  v (normal form): M=o, N=i, K=ci.  A = wv tiles, B = hn tiles.
//    D stored [b,c,j] coalesced -- exactly the layout P@V^T needs.
__global__ __launch_bounds__(256) void qkv_kernel(
    const bf16_t* __restrict__ hn,
    const bf16_t* __restrict__ wqb, const float* __restrict__ bq,
    const bf16_t* __restrict__ wkb, const float* __restrict__ bk,
    const bf16_t* __restrict__ wvb, const float* __restrict__ bv,
    bf16_t* __restrict__ qo, bf16_t* __restrict__ ko, bf16_t* __restrict__ vo) {
  int b = blockIdx.x >> 5;
  int blk = blockIdx.x & 31;
  int tid = threadIdx.x;
  int wid = tid >> 5, lane = tid & 31;
  int l15 = lane & 15, hi = lane >> 4;
  int i0 = blk * 128 + wid * 16;

  const float QSCALE = 1.4426950408889634f * 0.0625f;   // log2(e) * 256^-0.5

  const bf16_t* hb = hn + (size_t)b * N_SP * C_DIM;

  // A-fragments of hn (row m = i)
  const bf16_t* arow = hb + (size_t)(i0 + l15) * C_DIM + hi * 16;
  v16bf af[8];
#pragma unroll
  for (int kc = 0; kc < 8; kc++) af[kc] = load_frag_a(arow, kc * 32);

  bf16_t* qb = qo + (size_t)b * N_SP * C_DIM;
  bf16_t* kb = ko + (size_t)b * N_SP * C_DIM;
  for (int ot = 0; ot < 16; ot++) {
    const bf16_t* wqr = wqb + (size_t)(ot * 16 + l15) * C_DIM + hi * 16;
    const bf16_t* wkr = wkb + (size_t)(ot * 16 + l15) * C_DIM + hi * 16;
    v8f aq = {}; v8f ak = {};
#pragma unroll
    for (int kc = 0; kc < 8; kc++) {
      aq = wmma_bf16(af[kc], *(const v16bf*)(wqr + kc * 32), aq);
      ak = wmma_bf16(af[kc], *(const v16bf*)(wkr + kc * 32), ak);
    }
    int oc = ot * 16 + l15;            // D col n = o (per-lane)
    float bqv = bq[oc], bkv = bk[oc];
#pragma unroll
    for (int r = 0; r < 8; r++) {      // D row m = i0 + r + 8*hi
      int i = i0 + r + hi * 8;
      qb[(size_t)i * C_DIM + oc] = (bf16_t)((aq[r] + bqv) * QSCALE);
      kb[(size_t)i * C_DIM + oc] = (bf16_t)(ak[r] + bkv);
    }
  }

  // v: B-fragments of hn (col n = i, k = ci contiguous 32B per lane)
  v16bf bfh[8];
  const bf16_t* brow = hb + (size_t)(i0 + l15) * C_DIM + hi * 16;
#pragma unroll
  for (int kc = 0; kc < 8; kc++) bfh[kc] = *(const v16bf*)(brow + kc * 32);

  bf16_t* vb = vo + (size_t)b * C_DIM * N_SP;
  for (int ot = 0; ot < 16; ot++) {
    const bf16_t* wvr = wvb + (size_t)(ot * 16 + l15) * C_DIM + hi * 16;
    v8f av = {};
#pragma unroll
    for (int kc = 0; kc < 8; kc++)
      av = wmma_bf16(load_frag_a(wvr, kc * 32), bfh[kc], av);
#pragma unroll
    for (int r = 0; r < 8; r++) {      // D row m = out channel
      int c = ot * 16 + r + hi * 8;
      vb[(size_t)c * N_SP + i0 + l15] = (bf16_t)(av[r] + bv[c]);
    }
  }
}

// ---------------------------------------------------------------- flash attention
// Each wave: one 16-row Q strip (A-frags in VGPRs for whole kernel), streams
// 64-col K/V tiles.  K+V per batch = 4MB (all batches 32MB) -> L2-resident,
// so tiles are loaded straight from global (L2 hit) without LDS staging.
// Only LDS use: per-wave padded P buffer for the D-layout -> A-layout swap.
// S is already in log2-domain (scale folded into Q) -> p = exp2(S - m).
__global__ __launch_bounds__(256) void attn_kernel(
    const bf16_t* __restrict__ q, const bf16_t* __restrict__ k,
    const bf16_t* __restrict__ v, bf16_t* __restrict__ ob) {
  __shared__ __align__(16) bf16_t pbuf[8][16][72];   // 72 stride: bank-skewed
  int b = blockIdx.x >> 5;
  int blk = blockIdx.x & 31;
  int tid = threadIdx.x;
  int w = tid >> 5, lane = tid & 31;
  int l15 = lane & 15, hi = lane >> 4;
  int i0 = blk * 128 + w * 16;

  const bf16_t* qb = q + (size_t)b * N_SP * C_DIM;
  const bf16_t* kb = k + (size_t)b * N_SP * C_DIM;
  const bf16_t* vb = v + (size_t)b * C_DIM * N_SP;

  const bf16_t* qrow = qb + (size_t)(i0 + l15) * C_DIM + hi * 16;
  v16bf qf[8];
#pragma unroll
  for (int kc = 0; kc < 8; kc++) qf[kc] = load_frag_a(qrow, kc * 32);

  float row_m[8], row_l[8];
#pragma unroll
  for (int r = 0; r < 8; r++) { row_m[r] = -3.0e38f; row_l[r] = 0.f; }
  v8f o[16];
#pragma unroll
  for (int ct = 0; ct < 16; ct++) { v8f z = {}; o[ct] = z; }

  for (int j0 = 0; j0 < N_SP; j0 += 64) {
    if (j0 + 64 < N_SP) {                         // prefetch next K/V tile
      __builtin_prefetch(kb + (size_t)(j0 + 64 + l15) * C_DIM, 0, 1);
      __builtin_prefetch(vb + (size_t)l15 * N_SP + j0 + 64, 0, 1);
    }
    // S = Q^T K  (M=i, N=j, K=c): 4 n-subtiles x 8 k-steps
    v8f s[4];
#pragma unroll
    for (int ns = 0; ns < 4; ns++) { v8f z = {}; s[ns] = z; }
#pragma unroll
    for (int kc = 0; kc < 8; kc++) {
#pragma unroll
      for (int ns = 0; ns < 4; ns++) {
        const bf16_t* kp = kb + (size_t)(j0 + ns * 16 + l15) * C_DIM + kc * 32 + hi * 16;
        s[ns] = wmma_bf16(qf[kc], *(const v16bf*)kp, s[ns]);
      }
    }
    // online softmax (f32, log2 domain).  D layout: element r <-> row
    // i0+r+8*hi, col = lane&15; row reductions stay inside the 16-lane group.
    float mnew[8], corr[8];
#pragma unroll
    for (int r = 0; r < 8; r++) {
      float mx = fmaxf(fmaxf(s[0][r], s[1][r]), fmaxf(s[2][r], s[3][r]));
      mx = fmaxf(mx, __shfl_xor(mx, 1, 32));
      mx = fmaxf(mx, __shfl_xor(mx, 2, 32));
      mx = fmaxf(mx, __shfl_xor(mx, 4, 32));
      mx = fmaxf(mx, __shfl_xor(mx, 8, 32));
      mnew[r] = fmaxf(row_m[r], mx);
      corr[r] = exp2f(row_m[r] - mnew[r]);
      row_m[r] = mnew[r];
    }
#pragma unroll
    for (int ns = 0; ns < 4; ns++) {
#pragma unroll
      for (int r = 0; r < 8; r++) {
        float p = exp2f(s[ns][r] - mnew[r]);
        s[ns][r] = p;
        pbuf[w][r + hi * 8][ns * 16 + l15] = (bf16_t)p;   // D -> row-major LDS
      }
    }
#pragma unroll
    for (int r = 0; r < 8; r++) {
      float ls = s[0][r] + s[1][r] + s[2][r] + s[3][r];
      ls += __shfl_xor(ls, 1, 32);
      ls += __shfl_xor(ls, 2, 32);
      ls += __shfl_xor(ls, 4, 32);
      ls += __shfl_xor(ls, 8, 32);
      row_l[r] = row_l[r] * corr[r] + ls;
    }
#pragma unroll
    for (int ct = 0; ct < 16; ct++)
#pragma unroll
      for (int r = 0; r < 8; r++)
        o[ct][r] *= corr[r];

    // per-wave LDS round-trip: wait for P stores, then read A-fragments
    asm volatile("s_wait_dscnt 0" ::: "memory");
    v16bf pf[2];
#pragma unroll
    for (int kj = 0; kj < 2; kj++) {
      const bf16_t* pr = &pbuf[w][l15][kj * 32 + hi * 16];
      AB16 u;
      u.h[0] = *(const v8bf*)pr;
      u.h[1] = *(const v8bf*)(pr + 8);
      pf[kj] = u.v;
    }
    // O += P @ V^T  (M=i, N=c, K=j): B elem e -> j = j0 + kj*32 + hi*16 + e
#pragma unroll
    for (int ct = 0; ct < 16; ct++) {
#pragma unroll
      for (int kj = 0; kj < 2; kj++) {
        const bf16_t* vp = vb + (size_t)(ct * 16 + l15) * N_SP + j0 + kj * 32 + hi * 16;
        o[ct] = wmma_bf16(pf[kj], *(const v16bf*)vp, o[ct]);
      }
    }
  }

  bf16_t* obb = ob + (size_t)b * N_SP * C_DIM;
  float inv[8];
#pragma unroll
  for (int r = 0; r < 8; r++) inv[r] = 1.0f / row_l[r];
#pragma unroll
  for (int ct = 0; ct < 16; ct++) {
#pragma unroll
    for (int r = 0; r < 8; r++) {
      int i = i0 + r + hi * 8;
      obb[(size_t)i * C_DIM + ct * 16 + l15] = (bf16_t)(o[ct][r] * inv[r]);
    }
  }
}

// ---------------------------------------------------------------- out proj + residual
__global__ __launch_bounds__(256) void proj_kernel(
    const bf16_t* __restrict__ ob, const bf16_t* __restrict__ wob,
    const float* __restrict__ bo, const float* __restrict__ x,
    float* __restrict__ y) {
  int b = blockIdx.x >> 5;
  int blk = blockIdx.x & 31;
  int tid = threadIdx.x;
  int wid = tid >> 5, lane = tid & 31;
  int l15 = lane & 15, hi = lane >> 4;
  int i0 = blk * 128 + wid * 16;

  const bf16_t* obb = ob + (size_t)b * N_SP * C_DIM;
  const bf16_t* brow = obb + (size_t)(i0 + l15) * C_DIM + hi * 16;
  v16bf bfo[8];
#pragma unroll
  for (int kc = 0; kc < 8; kc++) bfo[kc] = *(const v16bf*)(brow + kc * 32);

  const float* xb = x + (size_t)b * C_DIM * N_SP;
  float* yb = y + (size_t)b * C_DIM * N_SP;
  for (int ot = 0; ot < 16; ot++) {
    const bf16_t* wor = wob + (size_t)(ot * 16 + l15) * C_DIM + hi * 16;
    v8f acc = {};
#pragma unroll
    for (int kc = 0; kc < 8; kc++)
      acc = wmma_bf16(load_frag_a(wor, kc * 32), bfo[kc], acc);
#pragma unroll
    for (int r = 0; r < 8; r++) {
      int c = ot * 16 + r + hi * 8;
      size_t idx = (size_t)c * N_SP + i0 + l15;
      yb[idx] = xb[idx] + bo[c] + acc[r];
    }
  }
}

// ---------------------------------------------------------------- launcher
extern "C" void kernel_launch(void* const* d_in, const int* in_sizes, int n_in,
                              void* d_out, int out_size, void* d_ws, size_t ws_size,
                              hipStream_t stream) {
  (void)in_sizes; (void)n_in; (void)out_size; (void)ws_size;
  const float* x   = (const float*)d_in[0];
  const float* gsc = (const float*)d_in[1];
  const float* gbi = (const float*)d_in[2];
  const float* wq  = (const float*)d_in[3];
  const float* bq  = (const float*)d_in[4];
  const float* wk  = (const float*)d_in[5];
  const float* bk  = (const float*)d_in[6];
  const float* wv  = (const float*)d_in[7];
  const float* bv  = (const float*)d_in[8];
  const float* wo  = (const float*)d_in[9];
  const float* bo  = (const float*)d_in[10];
  float* y = (float*)d_out;

  char* ws = (char*)d_ws;
  const size_t SZ = (size_t)8 * N_SP * C_DIM * sizeof(bf16_t);   // 16 MB each
  bf16_t* hn = (bf16_t*)(ws);            // GN output; reused as attn output
  bf16_t* qb = (bf16_t*)(ws + SZ);
  bf16_t* kb = (bf16_t*)(ws + 2 * SZ);
  bf16_t* vb = (bf16_t*)(ws + 3 * SZ);
  bf16_t* ob = hn;                       // hn dead after qkv_kernel -> alias
  bf16_t* wqb = (bf16_t*)(ws + 4 * SZ);
  bf16_t* wkb = wqb + 65536;
  bf16_t* wvb = wkb + 65536;
  bf16_t* wob = wvb + 65536;

  wcvt_kernel<<<256, 256, 0, stream>>>(wq, wqb, 65536);
  wcvt_kernel<<<256, 256, 0, stream>>>(wk, wkb, 65536);
  wcvt_kernel<<<256, 256, 0, stream>>>(wv, wvb, 65536);
  wcvt_kernel<<<256, 256, 0, stream>>>(wo, wob, 65536);
  gn_kernel<<<64, 256, 0, stream>>>(x, gsc, gbi, hn);
  qkv_kernel<<<256, 256, 0, stream>>>(hn, wqb, bq, wkb, bk, wvb, bv, qb, kb, vb);
  attn_kernel<<<256, 256, 0, stream>>>(qb, kb, vb, ob);
  proj_kernel<<<256, 256, 0, stream>>>(ob, wob, bo, x, y);
}
